// XLASDPAWrapper_23321672417335
// MI455X (gfx1250) — matrix operand order repaired and checked by reading, not compiled, not run on hardware
//
#include <hip/hip_runtime.h>
#include <hip/hip_bf16.h>

// ---------------------------------------------------------------------------
// Problem constants (from reference): B=1, S=2048, H=4096, NH=32, NKV=8, HD=128
// ---------------------------------------------------------------------------
#define S_LEN 2048
#define HDIM  4096
#define N_H   32
#define N_KV  8
#define HD_   128
#define ATTN_SCALE 0.08838834764831845f  // 128^-0.5

typedef __attribute__((ext_vector_type(16))) __bf16       v16bf;
typedef __attribute__((ext_vector_type(8)))  float        v8f;
typedef __attribute__((ext_vector_type(4)))  unsigned int u32x4;

union FragU { u32x4 u[2]; v16bf v; };

// ---------------------------------------------------------------------------
// CDNA5 async copy: global -> LDS, 16 bytes per lane, tracked by ASYNCcnt.
// LDS destination is the byte offset within the workgroup's LDS allocation
// (low 32 bits of the shared-aperture flat address, per ISA 10.2).
// ---------------------------------------------------------------------------
__device__ __forceinline__ void async_copy_b128(const void* gsrc, void* ldst) {
  unsigned lds_off = (unsigned)(size_t)ldst;
  asm volatile("global_load_async_to_lds_b128 %0, %1, off"
               :
               : "v"(lds_off), "v"(gsrc)
               : "memory");
}
__device__ __forceinline__ void async_wait0() {
  asm volatile("s_wait_asynccnt 0x0" ::: "memory");
}

// A-matrix fragment (16xK=32, bf16) per CDNA5 layout:
//   lanes 0-15: VGPR0-3 hold K=0..7, VGPR4-7 hold K=16..23
//   lanes 16-31: VGPR0-3 hold K=8..15, VGPR4-7 hold K=24..31
__device__ __forceinline__ v16bf frag_a32(const __bf16* base, int stride,
                                          int rowbase, int colbase) {
  const int lane = threadIdx.x & 31;
  const int m  = lane & 15;
  const int hf = lane >> 4;
  const __bf16* p = base + (size_t)(rowbase + m) * stride + colbase + hf * 8;
  FragU f;
  f.u[0] = *(const u32x4*)(p);
  f.u[1] = *(const u32x4*)(p + 16);
  return f.v;
}

// B-matrix fragment (K=32 x 16, bf16): lane n holds column n; lanes 0-15 hold
// K=0..7,16..23, lanes 16-31 hold K=8..15,24..31. Source memory holds each B *column* as a
// contiguous 32-element run (row `rowbase+n` of a row-major K-contiguous
// matrix), which is exactly what X@W^T and Q@K^T / P@V (via V^T) need.
__device__ __forceinline__ v16bf frag_b32(const __bf16* base, int stride,
                                          int rowbase, int colbase) {
  const int lane = threadIdx.x & 31;
  const int n  = lane & 15;
  const int hf = lane >> 4;
  const __bf16* p = base + (size_t)(rowbase + n) * stride + colbase + hf * 8;
  FragU f;
  f.u[0] = *(const u32x4*)(p);
  f.u[1] = *(const u32x4*)(p + 16);
  return f.v;
}

// ---------------------------------------------------------------------------
// fp32 -> bf16 conversion (grid-stride)
// ---------------------------------------------------------------------------
__global__ void f32_to_bf16(const float* __restrict__ in,
                            __bf16* __restrict__ out, long n) {
  long i = (long)blockIdx.x * blockDim.x + threadIdx.x;
  long stride = (long)gridDim.x * blockDim.x;
  for (; i < n; i += stride) out[i] = (__bf16)in[i];
}

// ---------------------------------------------------------------------------
// C[M,N] = A[M,K] @ W[N,K]^T   (bf16 in, fp32 accumulate, OutT out)
// Block: 256 threads (8 waves), tile 64(M) x 128(N), K-step 32, double-buffered
// LDS staged with global_load_async_to_lds_b128.
// ---------------------------------------------------------------------------
template <typename OutT>
__global__ __launch_bounds__(256) void gemm_xwt(const __bf16* __restrict__ A,
                                                const __bf16* __restrict__ W,
                                                OutT* __restrict__ C,
                                                int M, int N, int K) {
  constexpr int LDA = 40;  // 32 + 8 pad: keeps 16B alignment, skews banks
  constexpr int LDB = 40;
  __shared__ __align__(16) __bf16 sA[2][64 * LDA];
  __shared__ __align__(16) __bf16 sB[2][128 * LDB];

  const int tid   = threadIdx.x;
  const int mbase = blockIdx.y * 64;
  const int nbase = blockIdx.x * 128;
  const int w     = tid >> 5;
  const int waveM = w & 3;
  const int waveN = w >> 2;

  auto stage = [&](int buf, int k0) {
    {  // A tile: 64 rows x 32 k, one async b128 per thread
      int row = tid >> 2, col = (tid & 3) * 8;
      async_copy_b128(A + (size_t)(mbase + row) * K + k0 + col,
                      &sA[buf][row * LDA + col]);
    }
    {  // B tile: 128 rows x 32 k, two async b128 per thread
      int row = tid >> 1, col = (tid & 1) * 16;
      const __bf16* src = W + (size_t)(nbase + row) * K + k0 + col;
      async_copy_b128(src,     &sB[buf][row * LDB + col]);
      async_copy_b128(src + 8, &sB[buf][row * LDB + col + 8]);
    }
  };

  v8f acc[4] = {};

  stage(0, 0);
  async_wait0();
  __syncthreads();
  const int nk = K >> 5;
  for (int kb = 0; kb < nk; ++kb) {
    const int cur = kb & 1;
    if (kb + 1 < nk) stage(cur ^ 1, (kb + 1) << 5);

    v16bf a = frag_a32(&sA[cur][0], LDA, waveM * 16, 0);
#pragma unroll
    for (int nt = 0; nt < 4; ++nt) {
      v16bf b = frag_b32(&sB[cur][0], LDB, waveN * 64 + nt * 16, 0);
      acc[nt] = __builtin_amdgcn_wmma_f32_16x16x32_bf16(
          false, a, false, b, (short)0, acc[nt], false, false);
    }
    async_wait0();       // drain this wave's async copies into buf^1
    __syncthreads();     // publish buf^1 to all waves
  }

  // D layout: VGPR r, lanes 0-15 -> (M=r, N=lane), lanes 16-31 -> (M=r+8)
  const int lane = tid & 31, hf = lane >> 4, nin = lane & 15;
#pragma unroll
  for (int nt = 0; nt < 4; ++nt) {
#pragma unroll
    for (int r = 0; r < 8; ++r) {
      int m = mbase + waveM * 16 + r + hf * 8;
      int n = nbase + waveN * 64 + nt * 16 + nin;
      C[(size_t)m * N + n] = (OutT)acc[nt][r];
    }
  }
}

// ---------------------------------------------------------------------------
// In-place RoPE on bf16 activations laid out [S][nheads*HD]
// ---------------------------------------------------------------------------
__global__ void rope_bf16(__bf16* __restrict__ X,
                          const float* __restrict__ cosb,
                          const float* __restrict__ sinb, int nheads) {
  long idx = (long)blockIdx.x * blockDim.x + threadIdx.x;
  long total = (long)S_LEN * nheads * (HD_ / 2);
  if (idx >= total) return;
  int d = (int)(idx % (HD_ / 2));
  long t = idx / (HD_ / 2);
  int h = (int)(t % nheads);
  int s = (int)(t / nheads);
  __bf16* p = X + (size_t)s * nheads * HD_ + (size_t)h * HD_;
  float x0 = (float)p[d], x1 = (float)p[d + 64];
  float c0 = cosb[s * HD_ + d], c1 = cosb[s * HD_ + d + 64];
  float s0 = sinb[s * HD_ + d], s1 = sinb[s * HD_ + d + 64];
  p[d]      = (__bf16)(x0 * c0 - x1 * s0);   // rotate_half: -x2 for first half
  p[d + 64] = (__bf16)(x1 * c1 + x0 * s1);
}

// ---------------------------------------------------------------------------
// Causal flash attention. Grid (S/128, NH); block 256 threads = 8 waves.
// Wave w owns Q rows [qblk*128 + 16w, +15]; Q fragments stay in registers.
// Block loops over 32-key blocks; K staged via async copy, V staged transposed.
// ---------------------------------------------------------------------------
__global__ __launch_bounds__(256) void flash_attn(const __bf16* __restrict__ Q,
                                                  const __bf16* __restrict__ Kc,
                                                  const __bf16* __restrict__ Vc,
                                                  __bf16* __restrict__ O) {
  constexpr int LDK = 136;  // 128 + 8 pad
  constexpr int LDV = 40;   // 32 + 8 pad (rows of V^T hold 32 keys)
  constexpr int LDP = 40;
  __shared__ __align__(16) __bf16 sK[32 * LDK];
  __shared__ __align__(16) __bf16 sVT[HD_ * LDV];
  __shared__ __align__(16) __bf16 sP[8][16 * LDP];

  const int tid  = threadIdx.x;
  const int w    = tid >> 5;
  const int lane = tid & 31;
  const int hf   = lane >> 4;
  const int l16  = lane & 15;
  const int h    = blockIdx.y;
  const int kvh  = h >> 2;  // NH/NKV = 4
  const int qblk = blockIdx.x;
  const int qbase = qblk * 128 + w * 16;

  // Q fragments: 16 x 128 = 4 chunks of K=32
  v16bf qf[4];
#pragma unroll
  for (int c = 0; c < 4; ++c)
    qf[c] = frag_a32(Q + (size_t)h * HD_, N_H * HD_, qbase, c * 32);

  v8f acc[8] = {};
  float mstat[8], lstat[8];
#pragma unroll
  for (int r = 0; r < 8; ++r) { mstat[r] = -1e30f; lstat[r] = 0.0f; }

  const int nkb = qblk * 4 + 4;  // key blocks covering keys <= qblk*128+127
  for (int jb = 0; jb < nkb; ++jb) {
    const int jbase = jb * 32;
    {  // cooperative stage: K block 32x128 (async); V block -> V^T (manual)
      int row = tid >> 3, col = (tid & 7) * 16;
      const __bf16* ksrc =
          Kc + (size_t)(jbase + row) * (N_KV * HD_) + (size_t)kvh * HD_ + col;
      async_copy_b128(ksrc,     &sK[row * LDK + col]);
      async_copy_b128(ksrc + 8, &sK[row * LDK + col + 8]);

      const __bf16* vsrc =
          Vc + (size_t)(jbase + row) * (N_KV * HD_) + (size_t)kvh * HD_ + col;
      FragU vd;
      vd.u[0] = *(const u32x4*)(vsrc);
      vd.u[1] = *(const u32x4*)(vsrc + 8);
#pragma unroll
      for (int i = 0; i < 16; ++i) sVT[(col + i) * LDV + row] = vd.v[i];

      if (jb + 1 < nkb) {  // near-cache prefetch of the next K/V block
        __builtin_prefetch(ksrc + (size_t)32 * (N_KV * HD_), 0, 3);
        __builtin_prefetch(vsrc + (size_t)32 * (N_KV * HD_), 0, 3);
      }
    }
    async_wait0();
    __syncthreads();

    if (jbase <= qbase + 15) {  // else: fully masked for this wave's rows
      // S = Q K^T : two 16x16 tiles (keys jbase..+15 and jbase+16..+31)
      v8f s0 = {}, s1 = {};
#pragma unroll
      for (int c = 0; c < 4; ++c) {
        v16bf b0 = frag_b32(sK, LDK, 0, c * 32);
        v16bf b1 = frag_b32(sK, LDK, 16, c * 32);
        s0 = __builtin_amdgcn_wmma_f32_16x16x32_bf16(false, qf[c], false, b0,
                                                     (short)0, s0, false, false);
        s1 = __builtin_amdgcn_wmma_f32_16x16x32_bf16(false, qf[c], false, b1,
                                                     (short)0, s1, false, false);
      }

      float p0[8], p1[8];
#pragma unroll
      for (int r = 0; r < 8; ++r) {
        int qrow = qbase + r + hf * 8;
        p0[r] = (jbase + l16 <= qrow) ? s0[r] * ATTN_SCALE : -1e30f;
        p1[r] = (jbase + 16 + l16 <= qrow) ? s1[r] * ATTN_SCALE : -1e30f;
      }

      // online softmax, rows reduced across the 16 lanes of each half-wave
#pragma unroll
      for (int r = 0; r < 8; ++r) {
        float mx = fmaxf(p0[r], p1[r]);
#pragma unroll
        for (int off = 1; off < 16; off <<= 1)
          mx = fmaxf(mx, __shfl_xor(mx, off, 32));
        float mnew  = fmaxf(mstat[r], mx);
        float alpha = __expf(mstat[r] - mnew);
        p0[r] = __expf(p0[r] - mnew);
        p1[r] = __expf(p1[r] - mnew);
        float rs = p0[r] + p1[r];
#pragma unroll
        for (int off = 1; off < 16; off <<= 1)
          rs += __shfl_xor(rs, off, 32);
        lstat[r] = lstat[r] * alpha + rs;
        mstat[r] = mnew;
#pragma unroll
        for (int d = 0; d < 8; ++d) acc[d][r] = acc[d][r] * alpha;
      }

      // Reformat P from D-layout to A-layout via this wave's LDS scratch
      __bf16* myP = &sP[w][0];
#pragma unroll
      for (int r = 0; r < 8; ++r) {
        int row = r + hf * 8;
        myP[row * LDP + l16]      = (__bf16)p0[r];
        myP[row * LDP + 16 + l16] = (__bf16)p1[r];
      }
      v16bf pa = frag_a32(myP, LDP, 0, 0);

      // acc += P @ V  (B fragments read contiguously from V^T rows)
#pragma unroll
      for (int nt = 0; nt < 8; ++nt) {
        v16bf vb = frag_b32(sVT, LDV, nt * 16, 0);
        acc[nt] = __builtin_amdgcn_wmma_f32_16x16x32_bf16(
            false, pa, false, vb, (short)0, acc[nt], false, false);
      }
    }
    __syncthreads();
  }

  // epilogue: O[s, h*HD + d] = acc / l
#pragma unroll
  for (int nt = 0; nt < 8; ++nt) {
#pragma unroll
    for (int r = 0; r < 8; ++r) {
      int m = qbase + r + hf * 8;
      int d = h * HD_ + nt * 16 + l16;
      O[(size_t)m * (N_H * HD_) + d] = (__bf16)(acc[nt][r] / lstat[r]);
    }
  }
}

// ---------------------------------------------------------------------------
// Host-side orchestration
// ---------------------------------------------------------------------------
extern "C" void kernel_launch(void* const* d_in, const int* in_sizes, int n_in,
                              void* d_out, int out_size, void* d_ws,
                              size_t ws_size, hipStream_t stream) {
  (void)in_sizes; (void)n_in; (void)out_size; (void)ws_size;
  const float* X    = (const float*)d_in[0];
  const float* cosb = (const float*)d_in[1];
  const float* sinb = (const float*)d_in[2];
  const float* Wq   = (const float*)d_in[3];
  const float* Wk   = (const float*)d_in[4];
  const float* Wv   = (const float*)d_in[5];
  const float* Wo   = (const float*)d_in[6];
  float* out = (float*)d_out;

  char* ws = (char*)d_ws;
  size_t off = 0;
  auto alloc = [&](size_t bytes) -> char* {
    char* p = ws + off;
    off += (bytes + 255) & ~(size_t)255;
    return p;
  };
  __bf16* Xb  = (__bf16*)alloc((size_t)S_LEN * HDIM * 2);
  __bf16* Wqb = (__bf16*)alloc((size_t)HDIM * HDIM * 2);
  __bf16* Wkb = (__bf16*)alloc((size_t)N_KV * HD_ * HDIM * 2);
  __bf16* Wvb = (__bf16*)alloc((size_t)N_KV * HD_ * HDIM * 2);
  __bf16* Wob = (__bf16*)alloc((size_t)HDIM * HDIM * 2);
  __bf16* Qb  = (__bf16*)alloc((size_t)S_LEN * HDIM * 2);
  __bf16* Kb  = (__bf16*)alloc((size_t)S_LEN * N_KV * HD_ * 2);
  __bf16* Vb  = (__bf16*)alloc((size_t)S_LEN * N_KV * HD_ * 2);
  __bf16* Ab  = (__bf16*)alloc((size_t)S_LEN * HDIM * 2);

  auto cvt = [&](const float* src, __bf16* dst, long n) {
    long blocks = (n + 255) / 256;
    if (blocks > 4096) blocks = 4096;
    f32_to_bf16<<<dim3((unsigned)blocks), dim3(256), 0, stream>>>(src, dst, n);
  };
  cvt(X,  Xb,  (long)S_LEN * HDIM);
  cvt(Wq, Wqb, (long)HDIM * HDIM);
  cvt(Wk, Wkb, (long)N_KV * HD_ * HDIM);
  cvt(Wv, Wvb, (long)N_KV * HD_ * HDIM);
  cvt(Wo, Wob, (long)HDIM * HDIM);

  dim3 blk(256);
  // Q/K/V projections: C[M,N] = Xb[M,K] @ W[N,K]^T
  gemm_xwt<__bf16><<<dim3(HDIM / 128, S_LEN / 64), blk, 0, stream>>>(
      Xb, Wqb, Qb, S_LEN, HDIM, HDIM);
  gemm_xwt<__bf16><<<dim3((N_KV * HD_) / 128, S_LEN / 64), blk, 0, stream>>>(
      Xb, Wkb, Kb, S_LEN, N_KV * HD_, HDIM);
  gemm_xwt<__bf16><<<dim3((N_KV * HD_) / 128, S_LEN / 64), blk, 0, stream>>>(
      Xb, Wvb, Vb, S_LEN, N_KV * HD_, HDIM);

  // RoPE on Q and K
  long nq = (long)S_LEN * N_H * (HD_ / 2);
  rope_bf16<<<dim3((unsigned)((nq + 255) / 256)), dim3(256), 0, stream>>>(
      Qb, cosb, sinb, N_H);
  long nk = (long)S_LEN * N_KV * (HD_ / 2);
  rope_bf16<<<dim3((unsigned)((nk + 255) / 256)), dim3(256), 0, stream>>>(
      Kb, cosb, sinb, N_KV);

  // Causal flash attention
  flash_attn<<<dim3(S_LEN / 128, N_H), dim3(256), 0, stream>>>(Qb, Kb, Vb, Ab);

  // Output projection -> fp32 d_out
  gemm_xwt<float><<<dim3(HDIM / 128, S_LEN / 64), blk, 0, stream>>>(
      Ab, Wob, out, S_LEN, HDIM, HDIM);
}
